// SAGE_PRUNE_55224689492462
// MI455X (gfx1250) — matrix operand order repaired and checked by reading, not compiled, not run on hardware
//
#include <hip/hip_runtime.h>
#include <stdint.h>

#define F_INC 128
#define HIDC  256
#define OUTC  47
#define OUTP  48
#define N0C   292864
#define N1C   11264
#define BC    1024
#define E0C   281600
#define E1C   10240
#define HHIT  146432
#define PPC   2048

typedef __attribute__((ext_vector_type(16))) __bf16 v16bf;
typedef __attribute__((ext_vector_type(8)))  __bf16 v8bf;
typedef __attribute__((ext_vector_type(8)))  float  v8f;
typedef __attribute__((ext_vector_type(4)))  float  v4f;

union Frag16 { v16bf v; v8bf h[2]; };

__device__ __forceinline__ unsigned short f2bf(float f) {
    union { float f; unsigned u; } x; x.f = f;
    unsigned u = x.u;
    unsigned r = (u + 0x7FFFu + ((u >> 16) & 1u)) >> 16;
    return (unsigned short)r;
}

// ---------------- utility ----------------
__global__ void zero_f32_kernel(float* __restrict__ p, int n) {
    int i = blockIdx.x * blockDim.x + threadIdx.x;
    if (i < n) p[i] = 0.0f;
}

// node_src[pos] = feat_cache row (>=0) or ~(x row) (<0)
__global__ void build_node_src_kernel(const int* __restrict__ hit_idx,
                                      const int* __restrict__ hit_pos,
                                      const int* __restrict__ miss_pos,
                                      int* __restrict__ node_src) {
    int t = blockIdx.x * blockDim.x + threadIdx.x;
    if (t < HHIT) {
        node_src[hit_pos[t]] = hit_idx[t];
    } else if (t < N0C) {
        int i = t - HHIT;
        node_src[miss_pos[i]] = ~i;
    }
}

// ---------------- layer 0 edge aggregation: one wave per edge ----------------
// Source rows are ~single-use streams over ~144MB: load non-temporal so the
// L2-resident accumulator (5.8MB) and weights are not evicted.
__global__ __launch_bounds__(256)
void aggregate0_kernel(const int* __restrict__ esrc, const int* __restrict__ edst,
                       const int* __restrict__ node_src,
                       const float* __restrict__ feat_cache, const float* __restrict__ x,
                       float* __restrict__ agg, float* __restrict__ cnt) {
    int e = blockIdx.x * 8 + (threadIdx.x >> 5);
    int lane = threadIdx.x & 31;
    int s = esrc[e];
    int d = edst[e];
    int ns = node_src[s];
    const float* row = (ns >= 0) ? (feat_cache + (size_t)ns * F_INC)
                                 : (x + (size_t)(~ns) * F_INC);
    v4f v = __builtin_nontemporal_load(reinterpret_cast<const v4f*>(row) + lane);
    float* arow = agg + (size_t)d * F_INC + lane * 4;
    atomicAdd(arow + 0, v.x);
    atomicAdd(arow + 1, v.y);
    atomicAdd(arow + 2, v.z);
    atomicAdd(arow + 3, v.w);
    if (lane == 0) atomicAdd(cnt + d, 1.0f);
}

// mean0 (bf16) and dst rows 0..N1 (bf16)
__global__ __launch_bounds__(128)
void prep0_kernel(const float* __restrict__ agg, const float* __restrict__ cnt,
                  const int* __restrict__ node_src,
                  const float* __restrict__ feat_cache, const float* __restrict__ x,
                  unsigned short* __restrict__ mean_bf, unsigned short* __restrict__ adst_bf) {
    int r = blockIdx.x, c = threadIdx.x;
    float cn = fmaxf(cnt[r], 1.0f);
    mean_bf[(size_t)r * F_INC + c] = f2bf(agg[(size_t)r * F_INC + c] / cn);
    int ns = node_src[r];
    const float* row = (ns >= 0) ? (feat_cache + (size_t)ns * F_INC)
                                 : (x + (size_t)(~ns) * F_INC);
    adst_bf[(size_t)r * F_INC + c] = f2bf(row[c]);
}

// W0 (128,256) row-major -> transposed (N=256, K=128) bf16
__global__ void convert_w0_kernel(const float* __restrict__ Wl, const float* __restrict__ Wr,
                                  unsigned short* __restrict__ Wlt, unsigned short* __restrict__ Wrt) {
    int g = blockIdx.x * blockDim.x + threadIdx.x;
    if (g >= F_INC * HIDC) return;
    int k = g / HIDC, n = g % HIDC;
    Wlt[(size_t)n * F_INC + k] = f2bf(Wl[g]);
    Wrt[(size_t)n * F_INC + k] = f2bf(Wr[g]);
}

// W1 (256,47) row-major -> transposed+padded (N=48, K=256) bf16
__global__ void convert_w1_kernel(const float* __restrict__ Wl, const float* __restrict__ Wr,
                                  unsigned short* __restrict__ Wlt, unsigned short* __restrict__ Wrt) {
    int g = blockIdx.x * blockDim.x + threadIdx.x;
    if (g >= OUTP * HIDC) return;
    int n = g / HIDC, k = g % HIDC;
    unsigned short vl = 0, vr = 0;
    if (n < OUTC) {
        vl = f2bf(Wl[(size_t)k * OUTC + n]);
        vr = f2bf(Wr[(size_t)k * OUTC + n]);
    }
    Wlt[(size_t)n * HIDC + k] = vl;
    Wrt[(size_t)n * HIDC + k] = vr;
}

// ---------------- WMMA fragment loaders (contiguous k order, same for A and B) ----------------
// A 16x32 bf16: lane m=lane&15, half=lane>>4; element e -> K = half*16 + e
__device__ __forceinline__ v16bf load_fragA(const unsigned short* __restrict__ base,
                                            int m0, int kStride, int k0, int lane) {
    int m = lane & 15, half = lane >> 4;
    const unsigned short* p = base + (size_t)(m0 + m) * kStride + k0 + half * 16;
    Frag16 f;
    f.h[0] = *reinterpret_cast<const v8bf*>(p);
    f.h[1] = *reinterpret_cast<const v8bf*>(p + 8);
    return f.v;
}

// Async stage of one contiguous weight tile (nUShorts multiple of 8*blockDim)
// into LDS via ASYNCcnt-tracked copies.
__device__ __forceinline__ void async_stage_tile(const unsigned short* __restrict__ g,
                                                 unsigned short* sh, int nUShorts,
                                                 int tid, int nThreads) {
    for (int chunk = tid; chunk * 8 < nUShorts; chunk += nThreads) {
        unsigned loff = (unsigned)(size_t)&sh[chunk * 8];
        const unsigned short* gp = g + chunk * 8;
        asm volatile("global_load_async_to_lds_b128 %0, %1, off"
                     :: "v"(loff), "v"(gp) : "memory");
    }
}

// ---------------- GEMM layer 0: h1 = relu(mean@W0l + b0 + hdst@W0r), (N1,256) ----------------
// 4 waves / block share one column tile; B tiles staged once into LDS.
__global__ __launch_bounds__(128)
void gemm0_kernel(const unsigned short* __restrict__ Amean, const unsigned short* __restrict__ Adst,
                  const unsigned short* __restrict__ Bl, const unsigned short* __restrict__ Br,
                  const float* __restrict__ bias, float* __restrict__ out) {
    __shared__ __align__(32) unsigned short shBl[16 * F_INC];
    __shared__ __align__(32) unsigned short shBr[16 * F_INC];
    int tid = threadIdx.x;
    int wave = tid >> 5, lane = tid & 31;
    int tn = blockIdx.y;
    int tm = blockIdx.x * 4 + wave;

    async_stage_tile(Bl + (size_t)tn * 16 * F_INC, shBl, 16 * F_INC, tid, 128);
    async_stage_tile(Br + (size_t)tn * 16 * F_INC, shBr, 16 * F_INC, tid, 128);
    asm volatile("s_wait_asynccnt 0x0" ::: "memory");
    __syncthreads();

    int m = lane & 15, half = lane >> 4;
    v8f c = {0.f, 0.f, 0.f, 0.f, 0.f, 0.f, 0.f, 0.f};
#pragma unroll
    for (int k0 = 0; k0 < F_INC; k0 += 32) {
        v16bf a = load_fragA(Amean, tm * 16, F_INC, k0, lane);
        v16bf b = *reinterpret_cast<const v16bf*>(&shBl[m * F_INC + k0 + half * 16]);
        c = __builtin_amdgcn_wmma_f32_16x16x32_bf16(false, a, false, b, (short)0, c, false, false);
    }
#pragma unroll
    for (int k0 = 0; k0 < F_INC; k0 += 32) {
        v16bf a = load_fragA(Adst, tm * 16, F_INC, k0, lane);
        v16bf b = *reinterpret_cast<const v16bf*>(&shBr[m * F_INC + k0 + half * 16]);
        c = __builtin_amdgcn_wmma_f32_16x16x32_bf16(false, a, false, b, (short)0, c, false, false);
    }
    int col = tn * 16 + m;
    float bv = bias[col];
#pragma unroll
    for (int v = 0; v < 8; ++v) {
        int row = tm * 16 + v + 8 * half;
        out[(size_t)row * HIDC + col] = fmaxf(c[v] + bv, 0.0f);
    }
}

// ---------------- push/pull (no mutation of emb_cache input) ----------------
__global__ __launch_bounds__(256)
void pulls_resolve_kernel(const int* __restrict__ table, const int* __restrict__ pull_gid,
                          const int* __restrict__ push_gid, const int* __restrict__ push_bid,
                          const float* __restrict__ h1, const float* __restrict__ emb_cache,
                          float* __restrict__ pulled) {
    __shared__ int best;
    int i = blockIdx.x, t = threadIdx.x;
    if (t == 0) best = -1;
    __syncthreads();
    int slot = table[pull_gid[i]];
    int loc = -1;
    for (int j = t; j < PPC; j += 256)
        if (table[push_gid[j]] == slot) loc = j;   // strided ascending -> per-thread max
    if (loc >= 0) atomicMax(&best, loc);
    __syncthreads();
    float v = (best >= 0) ? h1[(size_t)push_bid[best] * HIDC + t]
                          : emb_cache[(size_t)slot * HIDC + t];
    pulled[(size_t)i * HIDC + t] = v;
}

// deterministic scatter: only the last pull targeting a given batch row writes
__global__ __launch_bounds__(256)
void pull_scatter_kernel(const int* __restrict__ pull_bid, const float* __restrict__ pulled,
                         float* __restrict__ h1) {
    __shared__ int dup;
    int i = blockIdx.x, t = threadIdx.x;
    if (t == 0) dup = 0;
    __syncthreads();
    int b = pull_bid[i];
    for (int j = i + 1 + t; j < PPC; j += 256)
        if (pull_bid[j] == b) dup = 1;
    __syncthreads();
    if (!dup) h1[(size_t)b * HIDC + t] = pulled[(size_t)i * HIDC + t];
}

__global__ __launch_bounds__(256)
void prep_adst1_kernel(const float* __restrict__ h1, unsigned short* __restrict__ adst) {
    int r = blockIdx.x, c = threadIdx.x;
    adst[(size_t)r * HIDC + c] = f2bf(h1[(size_t)r * HIDC + c]);
}

// ---------------- layer 1 edge aggregation: one wave per edge, 8 floats/lane ----------------
__global__ __launch_bounds__(256)
void aggregate1_kernel(const int* __restrict__ esrc, const int* __restrict__ edst,
                       const float* __restrict__ h1,
                       float* __restrict__ agg, float* __restrict__ cnt) {
    int e = blockIdx.x * 8 + (threadIdx.x >> 5);
    int lane = threadIdx.x & 31;
    int s = esrc[e], d = edst[e];
    const v4f* row = reinterpret_cast<const v4f*>(h1 + (size_t)s * HIDC);
    v4f v0 = __builtin_nontemporal_load(row + lane * 2 + 0);
    v4f v1 = __builtin_nontemporal_load(row + lane * 2 + 1);
    float* arow = agg + (size_t)d * HIDC + lane * 8;
    atomicAdd(arow + 0, v0.x); atomicAdd(arow + 1, v0.y);
    atomicAdd(arow + 2, v0.z); atomicAdd(arow + 3, v0.w);
    atomicAdd(arow + 4, v1.x); atomicAdd(arow + 5, v1.y);
    atomicAdd(arow + 6, v1.z); atomicAdd(arow + 7, v1.w);
    if (lane == 0) atomicAdd(cnt + d, 1.0f);
}

__global__ __launch_bounds__(256)
void prep_mean1_kernel(const float* __restrict__ agg, const float* __restrict__ cnt,
                       unsigned short* __restrict__ mean_bf) {
    int r = blockIdx.x, c = threadIdx.x;
    float cn = fmaxf(cnt[r], 1.0f);
    mean_bf[(size_t)r * HIDC + c] = f2bf(agg[(size_t)r * HIDC + c] / cn);
}

// ---------------- GEMM layer 1: logits = mean1@W1l + b1 + hdst1@W1r, (B,48 padded) ----------------
__global__ __launch_bounds__(128)
void gemm1_kernel(const unsigned short* __restrict__ Amean, const unsigned short* __restrict__ Adst,
                  const unsigned short* __restrict__ Bl, const unsigned short* __restrict__ Br,
                  const float* __restrict__ bias, float* __restrict__ out) {
    __shared__ __align__(32) unsigned short shBl[16 * HIDC];
    __shared__ __align__(32) unsigned short shBr[16 * HIDC];
    int tid = threadIdx.x;
    int wave = tid >> 5, lane = tid & 31;
    int tn = blockIdx.y;
    int tm = blockIdx.x * 4 + wave;

    async_stage_tile(Bl + (size_t)tn * 16 * HIDC, shBl, 16 * HIDC, tid, 128);
    async_stage_tile(Br + (size_t)tn * 16 * HIDC, shBr, 16 * HIDC, tid, 128);
    asm volatile("s_wait_asynccnt 0x0" ::: "memory");
    __syncthreads();

    int m = lane & 15, half = lane >> 4;
    v8f c = {0.f, 0.f, 0.f, 0.f, 0.f, 0.f, 0.f, 0.f};
#pragma unroll
    for (int k0 = 0; k0 < HIDC; k0 += 32) {
        v16bf a = load_fragA(Amean, tm * 16, HIDC, k0, lane);
        v16bf b = *reinterpret_cast<const v16bf*>(&shBl[m * HIDC + k0 + half * 16]);
        c = __builtin_amdgcn_wmma_f32_16x16x32_bf16(false, a, false, b, (short)0, c, false, false);
    }
#pragma unroll
    for (int k0 = 0; k0 < HIDC; k0 += 32) {
        v16bf a = load_fragA(Adst, tm * 16, HIDC, k0, lane);
        v16bf b = *reinterpret_cast<const v16bf*>(&shBr[m * HIDC + k0 + half * 16]);
        c = __builtin_amdgcn_wmma_f32_16x16x32_bf16(false, a, false, b, (short)0, c, false, false);
    }
    int col = tn * 16 + m;
    float bv = (col < OUTC) ? bias[col] : 0.0f;
#pragma unroll
    for (int v = 0; v < 8; ++v) {
        int row = tm * 16 + v + 8 * half;
        out[(size_t)row * OUTP + col] = c[v] + bv;
    }
}

// ---------------- log-softmax over 47 classes: one wave32 per row ----------------
__global__ __launch_bounds__(32)
void log_softmax_kernel(const float* __restrict__ logits, float* __restrict__ out) {
    int row = blockIdx.x, lane = threadIdx.x;
    float x0 = logits[(size_t)row * OUTP + lane];                 // lanes 0..31 < 47
    bool has2 = (lane + 32) < OUTC;
    float x1 = has2 ? logits[(size_t)row * OUTP + lane + 32] : -3.402823466e+38f;
    float mx = fmaxf(x0, x1);
#pragma unroll
    for (int o = 16; o >= 1; o >>= 1) mx = fmaxf(mx, __shfl_xor(mx, o, 32));
    float s = __expf(x0 - mx) + (has2 ? __expf(x1 - mx) : 0.0f);
#pragma unroll
    for (int o = 16; o >= 1; o >>= 1) s += __shfl_xor(s, o, 32);
    float ls = __logf(s);
    out[(size_t)row * OUTC + lane] = x0 - mx - ls;
    if (has2) out[(size_t)row * OUTC + lane + 32] = x1 - mx - ls;
}

extern "C" void kernel_launch(void* const* d_in, const int* in_sizes, int n_in,
                              void* d_out, int out_size, void* d_ws, size_t ws_size,
                              hipStream_t stream) {
    (void)in_sizes; (void)n_in; (void)out_size; (void)ws_size;
    const float* x          = (const float*)d_in[0];
    const float* feat_cache = (const float*)d_in[1];
    const float* emb_cache  = (const float*)d_in[2];
    const int*   hit_idx    = (const int*)d_in[3];
    const int*   hit_pos    = (const int*)d_in[4];
    const int*   miss_pos   = (const int*)d_in[5];
    const int*   ei0        = (const int*)d_in[6];   // (2, E0): row0=src, row1=dst
    const int*   ei1        = (const int*)d_in[7];   // (2, E1)
    const int*   table      = (const int*)d_in[8];
    const int*   push_bid   = (const int*)d_in[9];
    const int*   push_gid   = (const int*)d_in[10];
    const int*   pull_bid   = (const int*)d_in[11];
    const int*   pull_gid   = (const int*)d_in[12];
    const float* W0l        = (const float*)d_in[13];
    const float* b0         = (const float*)d_in[14];
    const float* W0r        = (const float*)d_in[15];
    const float* W1l        = (const float*)d_in[16];
    const float* b1         = (const float*)d_in[17];
    const float* W1r        = (const float*)d_in[18];
    float* out = (float*)d_out;

    char* ws = (char*)d_ws;
    size_t off = 0;
    auto alloc = [&](size_t bytes) -> char* {
        char* p = ws + off;
        off = (off + bytes + 255) & ~(size_t)255;
        return p;
    };

    int* node_src = (int*)alloc((size_t)N0C * 4);
    size_t zoff0 = off;
    float* agg0 = (float*)alloc((size_t)N1C * F_INC * 4);
    float* cnt0 = (float*)alloc((size_t)N1C * 4);
    float* agg1 = (float*)alloc((size_t)BC * HIDC * 4);
    float* cnt1 = (float*)alloc((size_t)BC * 4);
    size_t zoff1 = off;
    unsigned short* mean0 = (unsigned short*)alloc((size_t)N1C * F_INC * 2);
    unsigned short* adst0 = (unsigned short*)alloc((size_t)N1C * F_INC * 2);
    unsigned short* w0lt  = (unsigned short*)alloc((size_t)HIDC * F_INC * 2);
    unsigned short* w0rt  = (unsigned short*)alloc((size_t)HIDC * F_INC * 2);
    float* h1     = (float*)alloc((size_t)N1C * HIDC * 4);
    float* pulled = (float*)alloc((size_t)PPC * HIDC * 4);
    unsigned short* adst1 = (unsigned short*)alloc((size_t)BC * HIDC * 2);
    unsigned short* mean1 = (unsigned short*)alloc((size_t)BC * HIDC * 2);
    unsigned short* w1lt  = (unsigned short*)alloc((size_t)OUTP * HIDC * 2);
    unsigned short* w1rt  = (unsigned short*)alloc((size_t)OUTP * HIDC * 2);
    float* logits = (float*)alloc((size_t)BC * OUTP * 4);

    float* zbase = (float*)(ws + zoff0);
    int zcount = (int)((zoff1 - zoff0) / 4);

    zero_f32_kernel<<<(zcount + 255) / 256, 256, 0, stream>>>(zbase, zcount);
    build_node_src_kernel<<<(N0C + 255) / 256, 256, 0, stream>>>(hit_idx, hit_pos, miss_pos, node_src);
    aggregate0_kernel<<<E0C / 8, 256, 0, stream>>>(ei0, ei0 + E0C, node_src, feat_cache, x, agg0, cnt0);
    prep0_kernel<<<N1C, 128, 0, stream>>>(agg0, cnt0, node_src, feat_cache, x, mean0, adst0);
    convert_w0_kernel<<<(F_INC * HIDC + 255) / 256, 256, 0, stream>>>(W0l, W0r, w0lt, w0rt);
    gemm0_kernel<<<dim3(N1C / 64, HIDC / 16), 128, 0, stream>>>(mean0, adst0, w0lt, w0rt, b0, h1);
    pulls_resolve_kernel<<<PPC, 256, 0, stream>>>(table, pull_gid, push_gid, push_bid, h1, emb_cache, pulled);
    pull_scatter_kernel<<<PPC, 256, 0, stream>>>(pull_bid, pulled, h1);
    prep_adst1_kernel<<<BC, 256, 0, stream>>>(h1, adst1);
    aggregate1_kernel<<<E1C / 8, 256, 0, stream>>>(ei1, ei1 + E1C, h1, agg1, cnt1);
    prep_mean1_kernel<<<BC, 256, 0, stream>>>(agg1, cnt1, mean1);
    convert_w1_kernel<<<(OUTP * HIDC + 255) / 256, 256, 0, stream>>>(W1l, W1r, w1lt, w1rt);
    gemm1_kernel<<<dim3(BC / 64, OUTP / 16), 128, 0, stream>>>(mean1, adst1, w1lt, w1rt, b1, logits);
    log_softmax_kernel<<<BC, 32, 0, stream>>>(logits, out);
}
